// SingleStreamBlock_68015102099726
// MI455X (gfx1250) — matrix operand order repaired and checked by reading, not compiled, not run on hardware
//
#include <hip/hip_runtime.h>

// ---------------------------------------------------------------------------
// SingleStreamBlock for MI455X (gfx1250, wave32, WMMA).
// bf16 WMMA for all matmuls (f32 accumulate); fp32 for LN/RMS/softmax math.
// Weights are converted fp32->bf16 AND transposed to [n][k] so every LDS tile
// stage is a contiguous b128 copy; async global->LDS copies when available.
// ---------------------------------------------------------------------------

#define HID    3072
#define SEQ    2048
#define NHEAD  24
#define HDIM   128
#define MLP    12288
#define N1     21504      // 3*HID + MLP  (gemm1 output width)
#define QKVW   9216       // 3*HID
#define CATW   15360      // HID + MLP    (gemm2 input width)
#define LNEPS  1e-6f
#define RMSEPS 1.1920929e-07f

#define LDS_STRIDE 48     // bf16 elems per LDS tile row (96B: keeps b128 aligned)

typedef __bf16 bf16;
typedef __attribute__((ext_vector_type(8)))  bf16            bf16x8;
typedef __attribute__((ext_vector_type(16))) bf16            bf16x16;
typedef __attribute__((ext_vector_type(8)))  float           f32x8;
typedef __attribute__((ext_vector_type(8)))  unsigned short  u16x8;
typedef __attribute__((ext_vector_type(4)))  int             i32x4;

#define AS1 __attribute__((address_space(1)))
#define AS3 __attribute__((address_space(3)))

#if defined(__gfx1250__) && __has_builtin(__builtin_amdgcn_global_load_async_to_lds_b128)
#define HAVE_ASYNC_LDS 1
#endif

// 16-byte global -> LDS copy; async (ASYNCcnt-tracked, no VGPR round-trip) when
// the toolchain exposes the gfx1250 builtin, vector copy otherwise.
__device__ inline void copy8(unsigned short* l, const unsigned short* g) {
#ifdef HAVE_ASYNC_LDS
    __builtin_amdgcn_global_load_async_to_lds_b128((AS1 i32x4*)g, (AS3 i32x4*)l, 0, 0);
#else
    *(u16x8*)l = *(const u16x8*)g;
#endif
}

template <int N> __device__ inline void async_wait() {
#ifdef HAVE_ASYNC_LDS
#if __has_builtin(__builtin_amdgcn_s_wait_asynccnt)
    __builtin_amdgcn_s_wait_asynccnt(N);
#else
    asm volatile("s_wait_asynccnt %0" ::"i"(N) : "memory");
#endif
#endif
}

__device__ inline unsigned short f32_to_bf16_bits(float f) {
    union { float f; unsigned u; } v; v.f = f;
    unsigned u = v.u;
    unsigned r = (u + 0x7fffu + ((u >> 16) & 1u)) >> 16;   // round-nearest-even
    return (unsigned short)r;
}

__device__ inline float gelu_tanh(float v) {
    return 0.5f * v * (1.0f + tanhf(0.7978845608028654f * (v + 0.044715f * v * v * v)));
}

// ---------------------------------------------------------------------------
// WMMA fragment loaders (wave32, CDNA5 ISA 7.12.2 layouts).
// A (16x32 bf16): lane l (m=l&15, hi=l>>4) holds K=[8hi..8hi+7],[16+8hi..+7].
// B (32x16 bf16): lane l (n=l&15, hi=l>>4) holds K=[8hi..8hi+7],[16+8hi..+7] of col n
//                 -> LDS staged [col][k], k-contiguous.
// ---------------------------------------------------------------------------
__device__ inline bf16x16 frag_a(const unsigned short* p, int rstride, int row0, int kbase) {
    int lane = threadIdx.x & 31;
    int m = lane & 15, hi = lane >> 4;
    const unsigned short* q = p + (row0 + m) * rstride + kbase;
    bf16x8 lo  = *(const bf16x8*)(q + 8 * hi);
    bf16x8 hi8 = *(const bf16x8*)(q + 16 + 8 * hi);
    bf16x16 a;
#pragma unroll
    for (int i = 0; i < 8; i++) { a[i] = lo[i]; a[i + 8] = hi8[i]; }
    return a;
}

__device__ inline bf16x16 frag_b(const unsigned short* p, int rstride, int col0, int kbase) {
    int lane = threadIdx.x & 31;
    int n = lane & 15, hi = lane >> 4;
    const unsigned short* q = p + (col0 + n) * rstride + kbase;
    bf16x8 lo  = *(const bf16x8*)(q + 8 * hi);
    bf16x8 hi8 = *(const bf16x8*)(q + 16 + 8 * hi);
    bf16x16 b;
#pragma unroll
    for (int i = 0; i < 8; i++) { b[i] = lo[i]; b[i + 8] = hi8[i]; }
    return b;
}

__device__ inline float wred_sum32(float v) {
#pragma unroll
    for (int m = 16; m > 0; m >>= 1) v += __shfl_xor(v, m, 32);
    return v;
}

// ---------------------------------------------------------------------------
// 0) fused fp32->bf16 convert + transpose:  src f32 [K][N] -> dst bf16 [N][K]
// ---------------------------------------------------------------------------
__global__ __launch_bounds__(256) void k_convt(const float* __restrict__ src,
                                               unsigned short* __restrict__ dst,
                                               int K, int N) {
    __shared__ unsigned short tile[32][33];
    int tx = threadIdx.x & 31, ty = threadIdx.x >> 5;    // 32 x 8
    int n0 = blockIdx.x * 32, k0 = blockIdx.y * 32;
#pragma unroll
    for (int i = 0; i < 32; i += 8)
        tile[tx][ty + i] = f32_to_bf16_bits(src[(size_t)(k0 + ty + i) * N + n0 + tx]);
    __syncthreads();
#pragma unroll
    for (int i = 0; i < 32; i += 8)
        dst[(size_t)(n0 + ty + i) * K + k0 + tx] = tile[ty + i][tx];
}

// ---------------------------------------------------------------------------
// 1) mod = silu(vec) @ mod_w + mod_b          (1x3072)@(3072x9216)
// ---------------------------------------------------------------------------
__global__ __launch_bounds__(256) void k_mod(const float* __restrict__ vec,
                                             const float* __restrict__ mod_w,
                                             const float* __restrict__ mod_b,
                                             float* __restrict__ mod) {
    __shared__ float sv[HID];
    int t = threadIdx.x;
#pragma unroll
    for (int i = 0; i < HID / 256; i++) {
        float v = vec[t + i * 256];
        sv[t + i * 256] = v / (1.0f + __expf(-v));
    }
    __syncthreads();
    int j = blockIdx.x * 256 + t;
    float acc = 0.0f;
    for (int i = 0; i < HID; i++) acc += sv[i] * mod_w[(size_t)i * (3 * HID) + j];
    mod[j] = acc + mod_b[j];
}

// ---------------------------------------------------------------------------
// 2) x_mod = (1+scale)*LN(x) + shift  -> bf16
// ---------------------------------------------------------------------------
__global__ __launch_bounds__(256) void k_lnmod(const float* __restrict__ x,
                                               const float* __restrict__ mod,
                                               unsigned short* __restrict__ xmod) {
    __shared__ float sred[8];
    int row = blockIdx.x, t = threadIdx.x;
    int lane = t & 31, wave = t >> 5;
    float vals[HID / 256];
    float s = 0.0f;
#pragma unroll
    for (int i = 0; i < HID / 256; i++) {
        vals[i] = x[(size_t)row * HID + t + i * 256];
        s += vals[i];
    }
    s = wred_sum32(s);
    if (lane == 0) sred[wave] = s;
    __syncthreads();
    float tot = 0.0f;
#pragma unroll
    for (int w = 0; w < 8; w++) tot += sred[w];
    float mu = tot * (1.0f / HID);
    __syncthreads();
    float s2 = 0.0f;
#pragma unroll
    for (int i = 0; i < HID / 256; i++) { float d = vals[i] - mu; s2 += d * d; }
    s2 = wred_sum32(s2);
    if (lane == 0) sred[wave] = s2;
    __syncthreads();
    float tot2 = 0.0f;
#pragma unroll
    for (int w = 0; w < 8; w++) tot2 += sred[w];
    float rs = rsqrtf(tot2 * (1.0f / HID) + LNEPS);
#pragma unroll
    for (int i = 0; i < HID / 256; i++) {
        int c = t + i * 256;
        float y = (vals[i] - mu) * rs;
        float o = (1.0f + mod[HID + c]) * y + mod[c];
        xmod[(size_t)row * HID + c] = f32_to_bf16_bits(o);
    }
}

// ---------------------------------------------------------------------------
// GEMM core: C[64x128] per block, 8 waves (4m x 2n) each 16x64, K step 32,
// double-buffered LDS with async staging (3 b128 copies per thread per tile).
// A: bf16 [M][Kdim] row-major. Bt: bf16 [N][Kdim] (pre-transposed weights).
// ---------------------------------------------------------------------------
#define GEMM_BODY(Kdim, lda, A, Bt, EPILOGUE)                                          \
    __shared__ unsigned short lA[2][64 * LDS_STRIDE];                                  \
    __shared__ unsigned short lB[2][128 * LDS_STRIDE];                                 \
    const int t = threadIdx.x, lane = t & 31, wave = t >> 5;                           \
    const int wm = wave & 3, wn = wave >> 2;                                           \
    const int m0 = blockIdx.y * 64, n0 = blockIdx.x * 128;                             \
    f32x8 c[4] = {};                                                                   \
    auto stage = [&](unsigned short* bA, unsigned short* bB, int kb) {                 \
        {                                                                              \
            int row = t >> 2, k8 = (t & 3) * 8;                                        \
            copy8(&bA[row * LDS_STRIDE + k8], &A[(size_t)(m0 + row) * lda + kb + k8]); \
        }                                                                              \
        _Pragma("unroll") for (int i = 0; i < 2; i++) {                                \
            int ch = t + i * 256;                                                      \
            int n = ch >> 2, k8 = (ch & 3) * 8;                                        \
            copy8(&bB[n * LDS_STRIDE + k8], &Bt[(size_t)(n0 + n) * Kdim + kb + k8]);   \
        }                                                                              \
    };                                                                                 \
    stage(lA[0], lB[0], 0);                                                            \
    int buf = 0;                                                                       \
    for (int kb = 0; kb < Kdim; kb += 32, buf ^= 1) {                                  \
        bool hasnext = (kb + 32 < Kdim);                                               \
        if (hasnext) stage(lA[buf ^ 1], lB[buf ^ 1], kb + 32);                         \
        if (hasnext) async_wait<3>(); else async_wait<0>();                            \
        __syncthreads();                                                               \
        bf16x16 a = frag_a(lA[buf], LDS_STRIDE, wm * 16, 0);                           \
        _Pragma("unroll") for (int j = 0; j < 4; j++) {                                \
            bf16x16 b = frag_b(lB[buf], LDS_STRIDE, wn * 64 + j * 16, 0);              \
            c[j] = __builtin_amdgcn_wmma_f32_16x16x32_bf16(false, a, false, b,         \
                                                           (short)0, c[j], false,      \
                                                           false);                     \
        }                                                                              \
        __syncthreads();                                                               \
    }                                                                                  \
    const int hi = lane >> 4, n16 = lane & 15;                                         \
    _Pragma("unroll") for (int j = 0; j < 4; j++) {                                    \
        _Pragma("unroll") for (int r = 0; r < 8; r++) {                                \
            int row = m0 + wm * 16 + r + 8 * hi;                                       \
            int col = n0 + wn * 64 + j * 16 + n16;                                     \
            float v = c[j][r];                                                         \
            EPILOGUE;                                                                  \
        }                                                                              \
    }

// GEMM1: h = xmod @ w1 + b1 ; qkv cols -> fp32, mlp cols -> gelu -> bf16 cat
__global__ __launch_bounds__(256) void k_gemm1(const unsigned short* __restrict__ A,
                                               const unsigned short* __restrict__ Bt,
                                               const float* __restrict__ b1,
                                               float* __restrict__ qkv,
                                               unsigned short* __restrict__ cat) {
    GEMM_BODY(HID, HID, A, Bt, {
        v += b1[col];
        if (col < QKVW)
            qkv[(size_t)row * QKVW + col] = v;
        else
            cat[(size_t)row * CATW + HID + (col - QKVW)] = f32_to_bf16_bits(gelu_tanh(v));
    })
}

// GEMM2: out = x + gate * (cat @ w2 + b2)
__global__ __launch_bounds__(256) void k_gemm2(const unsigned short* __restrict__ A,
                                               const unsigned short* __restrict__ Bt,
                                               const float* __restrict__ b2,
                                               const float* __restrict__ x,
                                               const float* __restrict__ mod,
                                               float* __restrict__ out) {
    GEMM_BODY(CATW, CATW, A, Bt, {
        v += b2[col];
        out[(size_t)row * HID + col] = x[(size_t)row * HID + col] + mod[2 * HID + col] * v;
    })
}

// ---------------------------------------------------------------------------
// 4) per-(l,head): RMSNorm(q,k) + RoPE + bf16 pack (head-major [h][l][d])
// ---------------------------------------------------------------------------
__global__ __launch_bounds__(128) void k_qkvrope(const float* __restrict__ qkv,
                                                 const float* __restrict__ pe,
                                                 const float* __restrict__ qn_w,
                                                 const float* __restrict__ kn_w,
                                                 unsigned short* __restrict__ Qb,
                                                 unsigned short* __restrict__ Kb) {
    __shared__ float sred[2][4];
    __shared__ float qs[HDIM], ks[HDIM];
    int head = blockIdx.x, l = blockIdx.y, d = threadIdx.x;
    size_t base = (size_t)l * QKVW + head * HDIM + d;
    float q = qkv[base];
    float k = qkv[base + HID];
    float sq = q * q, sk = k * k;
#pragma unroll
    for (int m = 16; m > 0; m >>= 1) {
        sq += __shfl_xor(sq, m, 32);
        sk += __shfl_xor(sk, m, 32);
    }
    int wv = d >> 5;
    if ((d & 31) == 0) { sred[0][wv] = sq; sred[1][wv] = sk; }
    __syncthreads();
    float msq = (sred[0][0] + sred[0][1] + sred[0][2] + sred[0][3]) * (1.0f / HDIM);
    float msk = (sred[1][0] + sred[1][1] + sred[1][2] + sred[1][3]) * (1.0f / HDIM);
    qs[d] = q * rsqrtf(msq + RMSEPS) * qn_w[d];
    ks[d] = k * rsqrtf(msk + RMSEPS) * kn_w[d];
    __syncthreads();
    int p = d >> 1, i = d & 1;
    const float* pp = pe + (((size_t)l * 64 + p) * 2 + i) * 2;
    float qr = pp[0] * qs[2 * p] + pp[1] * qs[2 * p + 1];
    float kr = pp[0] * ks[2 * p] + pp[1] * ks[2 * p + 1];
    size_t ob = (size_t)head * SEQ * HDIM + (size_t)l * HDIM + d;
    Qb[ob] = f32_to_bf16_bits(qr);
    Kb[ob] = f32_to_bf16_bits(kr);
}

// ---------------------------------------------------------------------------
// 4b) V: convert + transpose per head: qkv f32 [l][.. v ..] -> Vt bf16 [h][d][l]
// ---------------------------------------------------------------------------
__global__ __launch_bounds__(256) void k_vtrans(const float* __restrict__ qkv,
                                                unsigned short* __restrict__ Vt) {
    __shared__ unsigned short tile[32][33];
    int tx = threadIdx.x & 31, ty = threadIdx.x >> 5;
    int l0 = blockIdx.x * 32, d0 = blockIdx.y * 32, head = blockIdx.z;
#pragma unroll
    for (int i = 0; i < 32; i += 8)
        tile[tx][ty + i] = f32_to_bf16_bits(
            qkv[(size_t)(l0 + ty + i) * QKVW + 2 * HID + head * HDIM + d0 + tx]);
    __syncthreads();
#pragma unroll
    for (int i = 0; i < 32; i += 8)
        Vt[(size_t)head * HDIM * SEQ + (size_t)(d0 + ty + i) * SEQ + l0 + tx] =
            tile[ty + i][tx];
}

// ---------------------------------------------------------------------------
// 5) Flash attention. Grid (qtile, head); 4 waves x 16 query rows. 32-key
//    tiles: S via 8 WMMAs, online softmax (f32), P through per-wave LDS,
//    P*V via 8 WMMAs. All staging = b128 (async when available).
// ---------------------------------------------------------------------------
__global__ __launch_bounds__(128) void k_attn(const unsigned short* __restrict__ Qb,
                                              const unsigned short* __restrict__ Kb,
                                              const unsigned short* __restrict__ Vt,
                                              unsigned short* __restrict__ cat) {
    __shared__ unsigned short lK[32 * 136];              // [key][d]
    __shared__ unsigned short lV[128 * LDS_STRIDE];      // [d][key]
    __shared__ unsigned short lP[4 * 16 * LDS_STRIDE];   // per-wave [m][key]
    const int t = threadIdx.x, lane = t & 31, wave = t >> 5;
    const int hi = lane >> 4, n16 = lane & 15;
    const int head = blockIdx.y;
    const int q0 = blockIdx.x * 64 + wave * 16;
    const float scale = 0.08838834764831845f;            // HDIM^-0.5
    const unsigned short* Qh  = Qb + (size_t)head * SEQ * HDIM;
    const unsigned short* Kh  = Kb + (size_t)head * SEQ * HDIM;
    const unsigned short* Vth = Vt + (size_t)head * HDIM * SEQ;

    bf16x16 qa[4];
#pragma unroll
    for (int kf = 0; kf < 4; kf++) {                     // resident Q frags
        const unsigned short* qp = Qh + (size_t)(q0 + n16) * HDIM + kf * 32;
        bf16x8 lo  = *(const bf16x8*)(qp + 8 * hi);
        bf16x8 hi8 = *(const bf16x8*)(qp + 16 + 8 * hi);
#pragma unroll
        for (int i = 0; i < 8; i++) { qa[kf][i] = lo[i]; qa[kf][i + 8] = hi8[i]; }
    }

    float mstate[8], lstate[8];
#pragma unroll
    for (int r = 0; r < 8; r++) { mstate[r] = -1e30f; lstate[r] = 0.0f; }
    f32x8 o[8] = {};

    for (int kb = 0; kb < SEQ; kb += 32) {
#pragma unroll
        for (int i = 0; i < 4; i++) {                    // K tile: [key][d]
            int ch = t + i * 128;
            int key = ch >> 4, d8 = (ch & 15) * 8;
            copy8(&lK[key * 136 + d8], &Kh[(size_t)(kb + key) * HDIM + d8]);
        }
#pragma unroll
        for (int i = 0; i < 4; i++) {                    // V tile: [d][key]
            int ch = t + i * 128;
            int d = ch >> 2, k8 = (ch & 3) * 8;
            copy8(&lV[d * LDS_STRIDE + k8], &Vth[(size_t)d * SEQ + kb + k8]);
        }
        async_wait<0>();
        __syncthreads();

        f32x8 s0 = {}, s1 = {};
#pragma unroll
        for (int kf = 0; kf < 4; kf++) {
            bf16x16 b0 = frag_b(lK, 136, 0, kf * 32);
            bf16x16 b1 = frag_b(lK, 136, 16, kf * 32);
            s0 = __builtin_amdgcn_wmma_f32_16x16x32_bf16(false, qa[kf], false, b0,
                                                         (short)0, s0, false, false);
            s1 = __builtin_amdgcn_wmma_f32_16x16x32_bf16(false, qa[kf], false, b1,
                                                         (short)0, s1, false, false);
        }

        float fac[8];
#pragma unroll
        for (int r = 0; r < 8; r++) {
            float v = fmaxf(s0[r], s1[r]);
#pragma unroll
            for (int m = 1; m < 16; m <<= 1) v = fmaxf(v, __shfl_xor(v, m, 32));
            float mn = fmaxf(mstate[r], v);
            fac[r] = __expf((mstate[r] - mn) * scale);
            mstate[r] = mn;
            float p0 = __expf((s0[r] - mn) * scale);
            float p1 = __expf((s1[r] - mn) * scale);
            s0[r] = p0; s1[r] = p1;
            float rs = p0 + p1;
#pragma unroll
            for (int m = 1; m < 16; m <<= 1) rs += __shfl_xor(rs, m, 32);
            lstate[r] = lstate[r] * fac[r] + rs;
        }
#pragma unroll
        for (int j = 0; j < 8; j++)
#pragma unroll
            for (int r = 0; r < 8; r++) o[j][r] = o[j][r] * fac[r];

        unsigned short* pw = lP + wave * 16 * LDS_STRIDE;   // C-frag -> A-frag
#pragma unroll
        for (int r = 0; r < 8; r++) {
            int m = r + 8 * hi;
            pw[m * LDS_STRIDE + n16]      = f32_to_bf16_bits(s0[r]);
            pw[m * LDS_STRIDE + 16 + n16] = f32_to_bf16_bits(s1[r]);
        }
        // same-wave LDS ops are in-order; lP region is wave-private
        bf16x16 pa = frag_a(pw, LDS_STRIDE, 0, 0);
#pragma unroll
        for (int j = 0; j < 8; j++) {
            bf16x16 vb = frag_b(lV, LDS_STRIDE, j * 16, 0);
            o[j] = __builtin_amdgcn_wmma_f32_16x16x32_bf16(false, pa, false, vb,
                                                           (short)0, o[j], false, false);
        }
        __syncthreads();
    }

#pragma unroll
    for (int j = 0; j < 8; j++) {
#pragma unroll
        for (int r = 0; r < 8; r++) {
            int row = q0 + r + 8 * hi;
            int d = j * 16 + n16;
            float v = o[j][r] / lstate[r];
            cat[(size_t)row * CATW + head * HDIM + d] = f32_to_bf16_bits(v);
        }
    }
}

// ---------------------------------------------------------------------------
extern "C" void kernel_launch(void* const* d_in, const int* in_sizes, int n_in,
                              void* d_out, int out_size, void* d_ws, size_t ws_size,
                              hipStream_t stream) {
    (void)in_sizes; (void)n_in; (void)out_size; (void)ws_size;
    const float* x     = (const float*)d_in[0];
    const float* vec   = (const float*)d_in[1];
    const float* pe    = (const float*)d_in[2];
    const float* w1    = (const float*)d_in[3];
    const float* b1    = (const float*)d_in[4];
    const float* w2    = (const float*)d_in[5];
    const float* b2    = (const float*)d_in[6];
    const float* mod_w = (const float*)d_in[7];
    const float* mod_b = (const float*)d_in[8];
    const float* qn_w  = (const float*)d_in[9];
    const float* kn_w  = (const float*)d_in[10];
    float* out = (float*)d_out;

    char* ws = (char*)d_ws;
    auto alloc = [&](size_t bytes) {
        char* p = ws;
        ws += (bytes + 255) & ~(size_t)255;
        return p;
    };
    float*          mod  = (float*)alloc((size_t)3 * HID * 4);
    unsigned short* xmod = (unsigned short*)alloc((size_t)SEQ * HID * 2);
    unsigned short* w1t  = (unsigned short*)alloc((size_t)HID * N1 * 2);   // [N1][HID]
    unsigned short* w2t  = (unsigned short*)alloc((size_t)CATW * HID * 2); // [HID][CATW]
    float*          qkv  = (float*)alloc((size_t)SEQ * QKVW * 4);
    unsigned short* cat  = (unsigned short*)alloc((size_t)SEQ * CATW * 2);
    unsigned short* Qb   = (unsigned short*)alloc((size_t)NHEAD * SEQ * HDIM * 2);
    unsigned short* Kb   = (unsigned short*)alloc((size_t)NHEAD * SEQ * HDIM * 2);
    unsigned short* Vt   = (unsigned short*)alloc((size_t)NHEAD * HDIM * SEQ * 2);

    k_convt<<<dim3(N1 / 32, HID / 32), 256, 0, stream>>>(w1, w1t, HID, N1);
    k_convt<<<dim3(HID / 32, CATW / 32), 256, 0, stream>>>(w2, w2t, CATW, HID);
    k_mod<<<(3 * HID) / 256, 256, 0, stream>>>(vec, mod_w, mod_b, mod);
    k_lnmod<<<SEQ, 256, 0, stream>>>(x, mod, xmod);
    k_gemm1<<<dim3(N1 / 128, SEQ / 64), 256, 0, stream>>>(xmod, w1t, b1, qkv, cat);
    k_qkvrope<<<dim3(NHEAD, SEQ), 128, 0, stream>>>(qkv, pe, qn_w, kn_w, Qb, Kb);
    k_vtrans<<<dim3(SEQ / 32, HDIM / 32, NHEAD), 256, 0, stream>>>(qkv, Vt);
    k_attn<<<dim3(SEQ / 64, NHEAD), 128, 0, stream>>>(Qb, Kb, Vt, cat);
    k_gemm2<<<dim3(HID / 128, SEQ / 64), 256, 0, stream>>>(cat, w2t, b2, x, mod, out);
}
